// TemporalAttention_15547781612051
// MI455X (gfx1250) — matrix operand order repaired and checked by reading, not compiled, not run on hardware
//
#include <hip/hip_runtime.h>
#include <math.h>

#define NTCOL 65536            // N * T = 2048 * 32
#define QK_SCALE 0.3535533905932738f   // 64^(-1/4)

typedef __attribute__((ext_vector_type(16))) __bf16 v16bf;
typedef __attribute__((ext_vector_type(8)))  float  v8f;
typedef __attribute__((ext_vector_type(8)))  unsigned int v8u;

__device__ __forceinline__ unsigned short f2bf(float f) {
  unsigned int u = __builtin_bit_cast(unsigned int, f);
  return (unsigned short)((u + 0x7FFFu + ((u >> 16) & 1u)) >> 16);
}
__device__ __forceinline__ float bf2f(unsigned short h) {
  unsigned int u = (unsigned int)h << 16;
  return __builtin_bit_cast(float, u);
}
// 16-bit A 16x32 layout: VGPR v, lane-half hi -> K pair base
__device__ __forceinline__ int a_k_of(int v, int hi) {
  return (v < 4 ? 0 : 16) + hi * 8 + ((v & 3) << 1);
}
// 16-bit B 32x16 layout: same K order as A (VGPR v, lane-half hi -> K pair base)
__device__ __forceinline__ int b_k_of(int v, int hi) { return (v < 4 ? 0 : 16) + hi * 8 + ((v & 3) << 1); }

// CDNA5 async DMA: copy 16B from global directly into LDS (ASYNCcnt-tracked).
__device__ __forceinline__ void async_load_b128(unsigned lds_addr, const void* gptr) {
  asm volatile("global_load_async_to_lds_b128 %0, %1, off"
               :: "v"(lds_addr), "v"(gptr) : "memory");
}
__device__ __forceinline__ void wait_asynccnt0() {
  asm volatile("s_wait_asynccnt 0x0" ::: "memory");
}

// ---------------- prep: weight conversion + rel-pos tables ----------------
__global__ void prep_kernel(const float* __restrict__ qkv_w, const float* __restrict__ proj_w,
                            const float* __restrict__ rpk, const float* __restrict__ rpv,
                            unsigned short* __restrict__ qkvw_bf, unsigned short* __restrict__ projw_bf,
                            unsigned short* __restrict__ krp, unsigned short* __restrict__ vrp,
                            float scale) {
  int stride = gridDim.x * blockDim.x;
  int g = blockIdx.x * blockDim.x + threadIdx.x;
  for (int i = g; i < 1536 * 512; i += stride) qkvw_bf[i] = f2bf(qkv_w[i]);
  for (int i = g; i < 512 * 512; i += stride)  projw_bf[i] = f2bf(proj_w[i]);
  for (int i = g; i < 32 * 32 * 64; i += stride) {
    int c = i & 63, ts = i >> 6, t = ts >> 5, s = ts & 31;
    int d = s - t + 64;  // |s-t| <= 31 < MAX_REL, no clipping needed
    krp[i] = f2bf(rpk[d * 64 + c] * scale);
    vrp[i] = f2bf(rpv[d * 64 + c]);
  }
}

// ---------------- GroupNorm stats: one thread per spatial position --------
__global__ __launch_bounds__(256) void gn_stats_kernel(const float* __restrict__ x,
                                                       float* __restrict__ meanb,
                                                       float* __restrict__ rstdb) {
  int s = blockIdx.x * 256 + threadIdx.x;   // 0..1023 (grid.x = 4)
  int g = blockIdx.y;                       // group 0..31
  int b = blockIdx.z;                       // batch 0..1
  size_t base = ((size_t)(b * 512 + g * 16) * 32) * 1024 + s;
  float sum = 0.f, sq = 0.f;
  for (int j = 0; j < 512; ++j) {           // 16 ch * 32 t
    float v = x[base + (size_t)j * 1024];
    sum += v; sq += v * v;
  }
  float m = sum * (1.0f / 512.0f);
  float var = fmaxf(sq * (1.0f / 512.0f) - m * m, 0.f);
  meanb[(b * 32 + g) * 1024 + s] = m;
  rstdb[(b * 32 + g) * 1024 + s] = rsqrtf(var + 1e-6f);
}

// ---------------- GroupNorm apply + transpose to [C, NT] bf16 -------------
__global__ void gn_apply_kernel(const float* __restrict__ x, const float* __restrict__ meanb,
                                const float* __restrict__ rstdb, const float* __restrict__ gamma,
                                const float* __restrict__ beta, unsigned short* __restrict__ xn) {
  int stride = gridDim.x * blockDim.x;
  for (int idx = blockIdx.x * blockDim.x + threadIdx.x; idx < (1 << 25); idx += stride) {
    int s = idx & 1023, t = (idx >> 10) & 31, c = (idx >> 15) & 511, b = idx >> 24;
    int g = c >> 4;
    float m = meanb[(b * 32 + g) * 1024 + s];
    float r = rstdb[(b * 32 + g) * 1024 + s];
    float v = (x[idx] - m) * r * gamma[c] + beta[c];
    xn[(size_t)c * NTCOL + (size_t)(b * 1024 + s) * 32 + t] = f2bf(v);
  }
}

// ---------------- BF16 WMMA GEMM: out[M,NT] = A[M,K] @ B[K,NT] + bias -----
// mode 0: fp32 out.  mode 1: bf16 out, rows < 1024 (q,k) scaled by qk_scale.
__global__ __launch_bounds__(256) void gemm_bf16_kernel(
    const unsigned short* __restrict__ A, const unsigned short* __restrict__ B,
    const float* __restrict__ bias, float* __restrict__ outF,
    unsigned short* __restrict__ outBF, int M, int K, int mode, float qk_scale) {
  // a_sh row stride 40 ushorts = 80B: 16B-aligned rows for async B128, and
  // r*20 % 64 hits 16 distinct banks for operand fetches.
  __shared__ __align__(16) unsigned short a_sh[128][40];   // [m][k]
  __shared__ __align__(16) unsigned short b_sh[128][34];   // [n][k] (transposed)
  int tid = threadIdx.x;
  int lane = tid & 31, wave = tid >> 5;
  int hi = lane >> 4, lo = lane & 15;
  int wm = wave & 1, wn = wave >> 1;     // 2x4 wave grid over 128x128 tile
  int m0 = blockIdx.y * 128, n0 = blockIdx.x * 128;

  v8f acc[4][2];
  for (int mt = 0; mt < 4; ++mt)
    for (int nt = 0; nt < 2; ++nt)
      for (int i = 0; i < 8; ++i) acc[mt][nt][i] = 0.f;

  for (int k0 = 0; k0 < K; k0 += 32) {
    // stage A tile via async DMA (global -> LDS, 16B chunks, no VGPR staging)
    for (int i = 0; i < 2; ++i) {
      int lin = i * 256 + tid;               // 512 b128 chunks (128 rows x 4)
      int r = lin >> 2, ch = lin & 3;
      unsigned lds = (unsigned)(size_t)&a_sh[r][ch * 8];
      const void* g = (const char*)A + ((size_t)(m0 + r) * K + k0 + ch * 8) * 2;
      async_load_b128(lds, g);
    }
    // prefetch next B k-slice into L2
    if (k0 + 32 < K)
      __builtin_prefetch((const char*)B + ((size_t)(k0 + 32) * NTCOL + n0 + 2 * tid) * 2, 0, 1);
    // stage B tile transposed: b_sh[n][k]
    for (int i = 0; i < 8; ++i) {
      int lin = i * 256 + tid;
      int kk = lin >> 6, n2 = lin & 63;
      unsigned int u = ((const unsigned int*)B)[(size_t)(k0 + kk) * (NTCOL >> 1) + (n0 >> 1) + n2];
      b_sh[2 * n2 + 0][kk] = (unsigned short)(u & 0xFFFFu);
      b_sh[2 * n2 + 1][kk] = (unsigned short)(u >> 16);
    }
    wait_asynccnt0();
    __syncthreads();

    v8u bu[2];
    for (int nt = 0; nt < 2; ++nt) {
      int nr = wn * 32 + nt * 16 + lo;
      for (int v = 0; v < 8; ++v)
        bu[nt][v] = *(const unsigned int*)&b_sh[nr][b_k_of(v, hi)];
    }
    for (int mt = 0; mt < 4; ++mt) {
      v8u au;
      int mr = wm * 64 + mt * 16 + lo;
      for (int v = 0; v < 8; ++v)
        au[v] = *(const unsigned int*)&a_sh[mr][a_k_of(v, hi)];
      v16bf av = __builtin_bit_cast(v16bf, au);
      for (int nt = 0; nt < 2; ++nt) {
        v16bf bv = __builtin_bit_cast(v16bf, bu[nt]);
        acc[mt][nt] = __builtin_amdgcn_wmma_f32_16x16x32_bf16(
            false, av, false, bv, (short)0, acc[mt][nt], false, false);
      }
    }
    __syncthreads();
  }

  for (int mt = 0; mt < 4; ++mt)
    for (int nt = 0; nt < 2; ++nt)
      for (int v = 0; v < 8; ++v) {
        int row = m0 + wm * 64 + mt * 16 + v + 8 * hi;
        int col = n0 + wn * 32 + nt * 16 + lo;
        float val = acc[mt][nt][v] + bias[row];
        if (mode == 1) {
          float sc = (row < 1024) ? qk_scale : 1.0f;   // scale q and k rows
          outBF[(size_t)row * NTCOL + col] = f2bf(val * sc);
        } else {
          outF[(size_t)row * NTCOL + col] = val;
        }
      }
}

// ---------------- attention: one wave per (n, head) -----------------------
__global__ __launch_bounds__(32) void attn_kernel(const unsigned short* __restrict__ qkv,
                                                  const unsigned short* __restrict__ krp,
                                                  const unsigned short* __restrict__ vrp,
                                                  unsigned short* __restrict__ aout) {
  __shared__ __align__(16) unsigned short q_sh[32][66];   // [t][c]  (q^T, pre-scaled)
  __shared__ __align__(16) unsigned short k_sh[32][66];   // [s][c]  (k^T, pre-scaled)
  __shared__ __align__(16) unsigned short v_sh[64][40];   // [c][s]  (80B rows, async)
  __shared__ __align__(16) float          w_sh[32][33];   // scores / probs fp32
  __shared__ __align__(16) unsigned short wb_sh[32][34];  // probs bf16
  __shared__ __align__(16) float          a_sh[32][66];   // out^T [t][c]

  int lane = threadIdx.x;
  int hi = lane >> 4, lo = lane & 15;
  int n = blockIdx.x >> 3, hd = blockIdx.x & 7;
  int col0 = n * 32;

  // v tile is a verbatim copy: async DMA straight into LDS (64 rows x 4 B128)
  for (int i = 0; i < 8; ++i) {
    int lin = i * 32 + lane;                 // 256 b128 chunks
    int c = lin >> 2, ch = lin & 3;
    unsigned lds = (unsigned)(size_t)&v_sh[c][ch * 8];
    const void* g = (const char*)qkv + ((size_t)(1024 + hd * 64 + c) * NTCOL + col0 + ch * 8) * 2;
    async_load_b128(lds, g);
  }
  // stage q,k transposed (manual: transpose can't be async-copied)
  for (int i = 0; i < 32; ++i) {
    int lin = i * 32 + lane;                 // 1024 dwords per matrix
    int c = lin >> 4, tp = (lin & 15) << 1;
    unsigned int uq = ((const unsigned int*)qkv)[(size_t)(hd * 64 + c) * (NTCOL >> 1) + (col0 >> 1) + (tp >> 1)];
    q_sh[tp][c] = (unsigned short)(uq & 0xFFFFu);
    q_sh[tp + 1][c] = (unsigned short)(uq >> 16);
    unsigned int uk = ((const unsigned int*)qkv)[(size_t)(512 + hd * 64 + c) * (NTCOL >> 1) + (col0 >> 1) + (tp >> 1)];
    k_sh[tp][c] = (unsigned short)(uk & 0xFFFFu);
    k_sh[tp + 1][c] = (unsigned short)(uk >> 16);
  }
  wait_asynccnt0();
  __syncthreads();

  // scores = qs^T ks : 2x2 tiles of 16x16, K = 64 (two x32 WMMAs)
  for (int ti = 0; ti < 2; ++ti)
    for (int si = 0; si < 2; ++si) {
      v8f accw; for (int i = 0; i < 8; ++i) accw[i] = 0.f;
      for (int kc = 0; kc < 64; kc += 32) {
        v8u au, bu;
        for (int v = 0; v < 8; ++v) {
          au[v] = *(const unsigned int*)&q_sh[ti * 16 + lo][kc + a_k_of(v, hi)];
          bu[v] = *(const unsigned int*)&k_sh[si * 16 + lo][kc + b_k_of(v, hi)];
        }
        accw = __builtin_amdgcn_wmma_f32_16x16x32_bf16(
            false, __builtin_bit_cast(v16bf, au), false, __builtin_bit_cast(v16bf, bu),
            (short)0, accw, false, false);
      }
      for (int v = 0; v < 8; ++v)
        w_sh[ti * 16 + v + 8 * hi][si * 16 + lo] = accw[v];
    }
  __syncthreads();

  // rel-k term + softmax: one row per lane
  {
    int t = lane;
    for (int s = 0; s < 32; ++s) {
      float acc2 = w_sh[t][s];
      const unsigned short* kr = krp + ((t * 32) + s) * 64;
      for (int c = 0; c < 64; ++c) acc2 += bf2f(q_sh[t][c]) * bf2f(kr[c]);
      w_sh[t][s] = acc2;
    }
    float mx = -1e30f;
    for (int s = 0; s < 32; ++s) mx = fmaxf(mx, w_sh[t][s]);
    float sum = 0.f;
    for (int s = 0; s < 32; ++s) { float e = __expf(w_sh[t][s] - mx); w_sh[t][s] = e; sum += e; }
    float inv = 1.0f / sum;
    for (int s = 0; s < 32; ++s) {
      float p = w_sh[t][s] * inv;
      w_sh[t][s] = p;
      wb_sh[t][s] = f2bf(p);
    }
  }
  __syncthreads();

  // a^T[t,c] = probs[t,s] @ v^T[s,c] : 2x4 tiles, K = 32
  for (int ti = 0; ti < 2; ++ti)
    for (int ci = 0; ci < 4; ++ci) {
      v8f acca; for (int i = 0; i < 8; ++i) acca[i] = 0.f;
      v8u au, bu;
      for (int v = 0; v < 8; ++v) {
        au[v] = *(const unsigned int*)&wb_sh[ti * 16 + lo][a_k_of(v, hi)];
        bu[v] = *(const unsigned int*)&v_sh[ci * 16 + lo][b_k_of(v, hi)];
      }
      acca = __builtin_amdgcn_wmma_f32_16x16x32_bf16(
          false, __builtin_bit_cast(v16bf, au), false, __builtin_bit_cast(v16bf, bu),
          (short)0, acca, false, false);
      for (int v = 0; v < 8; ++v)
        a_sh[ti * 16 + v + 8 * hi][ci * 16 + lo] = acca[v];
    }
  __syncthreads();

  // rel-v term
  for (int idx = lane; idx < 2048; idx += 32) {
    int t = idx >> 6, c = idx & 63;
    float s2 = a_sh[t][c];
    for (int s = 0; s < 32; ++s) s2 += w_sh[t][s] * bf2f(vrp[((t * 32) + s) * 64 + c]);
    a_sh[t][c] = s2;
  }
  __syncthreads();

  // store a as bf16 [512, NT]
  for (int i = 0; i < 32; ++i) {
    int lin = i * 32 + lane;
    int c = lin >> 4, tp = (lin & 15) << 1;
    unsigned int u = (unsigned int)f2bf(a_sh[tp][c]) | ((unsigned int)f2bf(a_sh[tp + 1][c]) << 16);
    ((unsigned int*)aout)[(size_t)(hd * 64 + c) * (NTCOL >> 1) + (col0 >> 1) + (tp >> 1)] = u;
  }
}

// ---------------- residual epilogue back to [b,c,t,h,w] -------------------
__global__ void final_kernel(const float* __restrict__ x, const float* __restrict__ y,
                             float* __restrict__ out) {
  int stride = gridDim.x * blockDim.x;
  for (int idx = blockIdx.x * blockDim.x + threadIdx.x; idx < (1 << 25); idx += stride) {
    int s = idx & 1023, t = (idx >> 10) & 31, c = (idx >> 15) & 511, b = idx >> 24;
    out[idx] = x[idx] + y[(size_t)c * NTCOL + (size_t)(b * 1024 + s) * 32 + t];
  }
}

extern "C" void kernel_launch(void* const* d_in, const int* in_sizes, int n_in,
                              void* d_out, int out_size, void* d_ws, size_t ws_size,
                              hipStream_t stream) {
  const float* x      = (const float*)d_in[0];
  const float* gamma  = (const float*)d_in[1];
  const float* beta   = (const float*)d_in[2];
  const float* qkv_w  = (const float*)d_in[3];
  const float* qkv_b  = (const float*)d_in[4];
  const float* rpk    = (const float*)d_in[5];
  const float* rpv    = (const float*)d_in[6];
  const float* proj_w = (const float*)d_in[7];
  const float* proj_b = (const float*)d_in[8];
  float* out = (float*)d_out;

  char* ws = (char*)d_ws;
  size_t off = 0;
  auto nxt = [&](size_t bytes) {
    char* p = ws + off;
    off = (off + bytes + 255) & ~(size_t)255;
    return p;
  };
  unsigned short* qkv_buf  = (unsigned short*)nxt((size_t)1536 * NTCOL * 2); // reused as y (fp32) later
  unsigned short* xn_buf   = (unsigned short*)nxt((size_t)512 * NTCOL * 2);
  unsigned short* a_buf    = (unsigned short*)nxt((size_t)512 * NTCOL * 2);
  unsigned short* qkvw_bf  = (unsigned short*)nxt((size_t)1536 * 512 * 2);
  unsigned short* projw_bf = (unsigned short*)nxt((size_t)512 * 512 * 2);
  unsigned short* krp      = (unsigned short*)nxt((size_t)32 * 32 * 64 * 2);
  unsigned short* vrp      = (unsigned short*)nxt((size_t)32 * 32 * 64 * 2);
  float* mean_buf          = (float*)nxt((size_t)2 * 32 * 1024 * 4);
  float* rstd_buf          = (float*)nxt((size_t)2 * 32 * 1024 * 4);
  float* y_buf = (float*)qkv_buf;  // qkv dead after attn; 128MB fits in 192MB region

  prep_kernel<<<2048, 256, 0, stream>>>(qkv_w, proj_w, rpk, rpv, qkvw_bf, projw_bf,
                                        krp, vrp, QK_SCALE);
  gn_stats_kernel<<<dim3(4, 32, 2), 256, 0, stream>>>(x, mean_buf, rstd_buf);
  gn_apply_kernel<<<8192, 256, 0, stream>>>(x, mean_buf, rstd_buf, gamma, beta, xn_buf);
  gemm_bf16_kernel<<<dim3(NTCOL / 128, 12), 256, 0, stream>>>(
      qkvw_bf, xn_buf, qkv_b, nullptr, qkv_buf, 1536, 512, 1, QK_SCALE);
  attn_kernel<<<65536, 32, 0, stream>>>(qkv_buf, krp, vrp, a_buf);
  gemm_bf16_kernel<<<dim3(NTCOL / 128, 4), 256, 0, stream>>>(
      projw_bf, a_buf, proj_b, y_buf, nullptr, 512, 512, 0, 1.0f);
  final_kernel<<<8192, 256, 0, stream>>>(x, y_buf, out);
}
